// StratifiedAttention_7378753815098
// MI455X (gfx1250) — compile-verified, matrix operand order checked by reading, not run on hardware
//
#include <hip/hip_runtime.h>

// ---------------------------------------------------------------------------
// StratifiedAttention for MI455X (gfx1250, wave32, WMMA).
// D=1024, H=16, HD=64, S=3, B=4, L=1024  -> M = B*L = 4096 tokens.
// Heavy GEMMs: bf16 WMMA (v_wmma_f32_16x16x32_bf16), f32 accum.
// Tile staging: GLOBAL_LOAD_ASYNC_TO_LDS_B128 (ASYNCcnt) with double-buffered
// LDS when the toolchain exposes the builtin; sync fallback otherwise.
// ---------------------------------------------------------------------------

typedef __attribute__((ext_vector_type(16))) __bf16 v16bf;
typedef __attribute__((ext_vector_type(8)))  float  v8f;
typedef int v4i __attribute__((vector_size(16)));
typedef v4i __attribute__((address_space(1)))* gv4i_p;   // global int4*
typedef v4i __attribute__((address_space(3)))* lv4i_p;   // LDS int4*

#define NT 4096      // tokens
#define DM 1024      // d_model
#define NH 16
#define HD 64
#define NS 3

#if __has_builtin(__builtin_amdgcn_global_load_async_to_lds_b128)
#define HAVE_ASYNC_LDS 1
#endif

__device__ __forceinline__ unsigned f2bfu(float f) {
  unsigned u = __builtin_bit_cast(unsigned, f);
  unsigned r = u + 0x7FFFu + ((u >> 16) & 1u);   // round-to-nearest-even
  return r >> 16;
}
__device__ __forceinline__ float bf2f(unsigned short s) {
  return __builtin_bit_cast(float, ((unsigned)s) << 16);
}

// 16-byte global -> LDS copy; async (no VGPR data path) when available.
__device__ __forceinline__ void async_cp16(const void* g, void* l) {
#ifdef HAVE_ASYNC_LDS
  __builtin_amdgcn_global_load_async_to_lds_b128((gv4i_p)g, (lv4i_p)l, 0, 0);
#else
  *(uint4*)l = *(const uint4*)g;
#endif
}
__device__ __forceinline__ void wait_async() {
#ifdef HAVE_ASYNC_LDS
#if __has_builtin(__builtin_amdgcn_s_wait_asynccnt)
  __builtin_amdgcn_s_wait_asynccnt(0);
#else
  asm volatile("s_wait_asynccnt 0" ::: "memory");
#endif
#endif
}

// ---------------------------------------------------------------------------
// fp32 -> bf16 bulk convert (vectorized, n multiple of 4)
// ---------------------------------------------------------------------------
__global__ __launch_bounds__(256)
void cvt_f32_bf16_kernel(const float* __restrict__ in,
                         unsigned short* __restrict__ out, int n4) {
  int i = blockIdx.x * 256 + threadIdx.x;
  if (i < n4) {
    float4 v = ((const float4*)in)[i];
    uint2 pk;
    pk.x = f2bfu(v.x) | (f2bfu(v.y) << 16);
    pk.y = f2bfu(v.z) | (f2bfu(v.w) << 16);
    ((uint2*)out)[i] = pk;
  }
}

// ---------------------------------------------------------------------------
// Generic tiled WMMA GEMM:  C[M,N] = A[M,K](bf16) @ W[N,K]^T(bf16) + bias(f32)
// Block 256 threads = 8 waves (4 x 2), tile BM=128 BN=128 BK=32.
// Wave owns a 32x64 C-subtile = 2x4 wmma accumulators.
// Double-buffered LDS staging via async-to-LDS.
// ---------------------------------------------------------------------------
#define MODE_BF16         0
#define MODE_BF16_RELU    1
#define MODE_F32_SET_MASK 2
#define MODE_F32_ACC_MASK 3
#define MODE_F32          4

constexpr int BM = 128, BN = 128, BK = 32, LDT = BK + 8;  // pad vs bank conflicts

template<bool AMASK>
__device__ __forceinline__ void stage_tiles(
    const unsigned short* __restrict__ A, const unsigned short* __restrict__ W,
    unsigned short* As, unsigned short* Bs,
    int m0, int n0, int K, int kb,
    const int* __restrict__ assign, int stratum, int tid) {
#pragma unroll
  for (int i = 0; i < 2; i++) {                 // A tile: 512 x 16B
    int idx = tid + i * 256;
    int row = idx >> 2;
    int c8  = (idx & 3) * 8;
    const unsigned short* g = A + (size_t)(m0 + row) * K + kb + c8;
    unsigned short* l = &As[row * LDT + c8];
    if (AMASK) {
      uint4 val = *(const uint4*)g;             // sync path: mask in registers
      if (assign[m0 + row] != stratum) val = make_uint4(0u, 0u, 0u, 0u);
      *(uint4*)l = val;
    } else {
      async_cp16(g, l);
    }
  }
#pragma unroll
  for (int i = 0; i < 2; i++) {                 // B tile: 512 x 16B
    int idx = tid + i * 256;
    int row = idx >> 2;
    int c8  = (idx & 3) * 8;
    async_cp16(W + (size_t)(n0 + row) * K + kb + c8, &Bs[row * LDT + c8]);
  }
}

template<int OUTMODE, bool AMASK>
__global__ __launch_bounds__(256)
void gemm_bf16_wmma(const unsigned short* __restrict__ A,
                    const unsigned short* __restrict__ W,   // bf16 [N,K]
                    const float* __restrict__ bias,
                    void* __restrict__ Cout,
                    const int* __restrict__ assign,
                    int stratum, int M, int N, int K) {
  __shared__ unsigned short As[2][BM * LDT];
  __shared__ unsigned short Bs[2][BN * LDT];

  const int tid  = threadIdx.x;
  const int lane = tid & 31;
  const int wave = tid >> 5;
  const int wm = wave >> 1, wn = wave & 1;
  const int m16 = lane & 15, hi = lane >> 4, kb8 = hi * 8;

  const int n0 = blockIdx.x * BN;
  const int m0 = blockIdx.y * BM;

  const v8f vzero = {0.f, 0.f, 0.f, 0.f, 0.f, 0.f, 0.f, 0.f};
  v8f acc[2][4];
#pragma unroll
  for (int i = 0; i < 2; i++)
#pragma unroll
    for (int j = 0; j < 4; j++) acc[i][j] = vzero;

  union Frag { v16bf v; uint4 q[2]; };

  stage_tiles<AMASK>(A, W, As[0], Bs[0], m0, n0, K, 0, assign, stratum, tid);
  wait_async();
  __syncthreads();

  int cur = 0;
  for (int kb = 0; kb < K; kb += BK) {
    if (kb + BK < K) {
      stage_tiles<AMASK>(A, W, As[cur ^ 1], Bs[cur ^ 1], m0, n0, K, kb + BK,
                         assign, stratum, tid);
      if (kb + 2 * BK < K)   // L2 prefetch of the tile after next (weights)
        __builtin_prefetch(W + (size_t)(n0 + (tid >> 1)) * K + kb + 2 * BK, 0, 1);
    }

    Frag bF[4], aF[2];
#pragma unroll
    for (int nf = 0; nf < 4; nf++) {
      const unsigned short* p = &Bs[cur][(wn * 64 + nf * 16 + m16) * LDT + kb8];
      bF[nf].q[0] = *(const uint4*)p;
      bF[nf].q[1] = *(const uint4*)(p + 16);
    }
#pragma unroll
    for (int mf = 0; mf < 2; mf++) {
      const unsigned short* p = &As[cur][(wm * 32 + mf * 16 + m16) * LDT + kb8];
      aF[mf].q[0] = *(const uint4*)p;
      aF[mf].q[1] = *(const uint4*)(p + 16);
    }
#pragma unroll
    for (int mf = 0; mf < 2; mf++)
#pragma unroll
      for (int nf = 0; nf < 4; nf++)
        acc[mf][nf] = __builtin_amdgcn_wmma_f32_16x16x32_bf16(
            false, aF[mf].v, false, bF[nf].v, (short)0, acc[mf][nf], false, false);

    wait_async();          // next buffer fully landed in LDS
    __syncthreads();       // everyone done reading cur
    cur ^= 1;
  }

  // --- epilogue: C layout lane=N, reg=M (lanes 16-31 -> M+8) ---
#pragma unroll
  for (int mf = 0; mf < 2; mf++) {
#pragma unroll
    for (int nf = 0; nf < 4; nf++) {
      int col = n0 + wn * 64 + nf * 16 + m16;
      float bv = bias[col];
#pragma unroll
      for (int r = 0; r < 8; r++) {
        int row = m0 + wm * 32 + mf * 16 + r + 8 * hi;
        float v = acc[mf][nf][r] + bv;
        if constexpr (OUTMODE == MODE_BF16_RELU) v = fmaxf(v, 0.0f);
        if constexpr (OUTMODE == MODE_BF16 || OUTMODE == MODE_BF16_RELU) {
          ((unsigned short*)Cout)[(size_t)row * N + col] = (unsigned short)f2bfu(v);
        } else if constexpr (OUTMODE == MODE_F32_SET_MASK) {
          float mk = (assign[row] == stratum) ? 1.0f : 0.0f;
          ((float*)Cout)[(size_t)row * N + col] = mk * v;
        } else if constexpr (OUTMODE == MODE_F32_ACC_MASK) {
          float mk = (assign[row] == stratum) ? 1.0f : 0.0f;
          ((float*)Cout)[(size_t)row * N + col] += mk * v;
        } else {
          ((float*)Cout)[(size_t)row * N + col] = v;
        }
      }
    }
  }
}

// ---------------------------------------------------------------------------
// Router second layer + argmax dispatch (tiny; argmax(softmax)==argmax).
// ---------------------------------------------------------------------------
__global__ __launch_bounds__(256)
void router_logits_kernel(const unsigned short* __restrict__ h,   // [4096][512] bf16
                          const float* __restrict__ w2,           // [3][512]
                          const float* __restrict__ b2,           // [3]
                          int* __restrict__ assign,
                          int* __restrict__ out_assign) {
  int t = blockIdx.x * 256 + threadIdx.x;
  if (t >= NT) return;
  const unsigned short* hr = h + (size_t)t * 512;
  float a0 = 0.f, a1 = 0.f, a2 = 0.f;
  for (int i = 0; i < 512; i++) {
    float hv = bf2f(hr[i]);
    a0 += hv * w2[i];
    a1 += hv * w2[512 + i];
    a2 += hv * w2[1024 + i];
  }
  a0 += b2[0]; a1 += b2[1]; a2 += b2[2];
  int am = 0; float mv = a0;
  if (a1 > mv) { mv = a1; am = 1; }
  if (a2 > mv) { mv = a2; am = 2; }
  assign[t] = am;
  out_assign[t] = am;
}

// ---------------------------------------------------------------------------
// Flash-style WMMA attention for one stratum.
// grid = (B*H, L/128); block = 256 (8 waves). Wave owns 16 q-rows.
// Streams 32-key tiles: S = Q K^T (4 wmma), online softmax (f32, cross-lane
// reduce over 16-lane groups), P restaged via per-wave LDS to A-layout,
// O += P V (4 wmma). K tile staged with async-to-LDS.
// ---------------------------------------------------------------------------
__global__ __launch_bounds__(256)
void attn_wmma_kernel(const unsigned short* __restrict__ qkv,  // [4096][3072] bf16
                      unsigned short* __restrict__ Obf) {      // [4096][1024] bf16
  const int bh = blockIdx.x;          // 0..63
  const int b  = bh >> 4;
  const int h  = bh & 15;
  const int qt = blockIdx.y;          // 0..7 -> 128 q rows
  const int wave = threadIdx.x >> 5;
  const int lane = threadIdx.x & 31;
  const int m16 = lane & 15, hi = lane >> 4, kb8 = hi * 8;

  __shared__ unsigned short Ks[32][72];       // [key][d]
  __shared__ unsigned short Vs[64][40];       // [d][key]  transposed
  __shared__ unsigned short Ps[8][16][40];    // per-wave P staging [qrow][key]

  union Frag { v16bf v; uint4 q[2]; };
  const v8f vzero = {0.f, 0.f, 0.f, 0.f, 0.f, 0.f, 0.f, 0.f};

  const int qrow = qt * 128 + wave * 16;
  const size_t tqbase = (size_t)(b * 1024 + qrow + m16) * 3072 + h * HD;

  Frag qa[2];
#pragma unroll
  for (int f = 0; f < 2; f++) {
    const unsigned short* p = qkv + tqbase + f * 32 + kb8;
    qa[f].q[0] = *(const uint4*)p;
    qa[f].q[1] = *(const uint4*)(p + 16);
  }

  v8f o[4];
#pragma unroll
  for (int f = 0; f < 4; f++) o[f] = vzero;
  float rm[8], rl[8];
#pragma unroll
  for (int r = 0; r < 8; r++) { rm[r] = -1e30f; rl[r] = 0.f; }

  for (int kt = 0; kt < 32; kt++) {
    {
      int kk = threadIdx.x >> 3;
      int c8 = (threadIdx.x & 7) * 8;
      const size_t tk = (size_t)(b * 1024 + kt * 32 + kk) * 3072 + h * HD + c8;
      async_cp16(qkv + tk + 1024, &Ks[kk][c8]);       // K tile: pure copy
      union { uint4 q; unsigned short s[8]; } vv;     // V tile: transpose store
      vv.q = *(const uint4*)(qkv + tk + 2048);
#pragma unroll
      for (int j = 0; j < 8; j++) Vs[c8 + j][kk] = vv.s[j];
    }
    wait_async();
    __syncthreads();

    // --- scores: S[16 x 32] = Q(16x64) . K^T ---
    v8f sc[2];
#pragma unroll
    for (int nf = 0; nf < 2; nf++) {
      sc[nf] = vzero;
#pragma unroll
      for (int kc = 0; kc < 2; kc++) {
        Frag kf;
        const unsigned short* p = &Ks[nf * 16 + m16][kc * 32 + kb8];
        kf.q[0] = *(const uint4*)p;
        kf.q[1] = *(const uint4*)(p + 16);
        sc[nf] = __builtin_amdgcn_wmma_f32_16x16x32_bf16(
            false, qa[kc].v, false, kf.v, (short)0, sc[nf], false, false);
      }
    }

    // --- online softmax ---
    float alpha[8];
#pragma unroll
    for (int r = 0; r < 8; r++) {
      float v = fmaxf(sc[0][r], sc[1][r]) * 0.125f;
#pragma unroll
      for (int mk = 1; mk < 16; mk <<= 1) v = fmaxf(v, __shfl_xor(v, mk, 32));
      float nm = fmaxf(rm[r], v);
      alpha[r] = __expf(rm[r] - nm);
      float p0 = __expf(sc[0][r] * 0.125f - nm);
      float p1 = __expf(sc[1][r] * 0.125f - nm);
      Ps[wave][r + 8 * hi][m16]      = (unsigned short)f2bfu(p0);
      Ps[wave][r + 8 * hi][16 + m16] = (unsigned short)f2bfu(p1);
      float ps = p0 + p1;
#pragma unroll
      for (int mk = 1; mk < 16; mk <<= 1) ps += __shfl_xor(ps, mk, 32);
      rl[r] = rl[r] * alpha[r] + ps;
      rm[r] = nm;
    }
#pragma unroll
    for (int f = 0; f < 4; f++)
#pragma unroll
      for (int r = 0; r < 8; r++) o[f][r] *= alpha[r];

    // --- P as A-fragment (wave-private LDS; DS ops in-order per wave) ---
    Frag pa;
    {
      const unsigned short* pp = &Ps[wave][m16][kb8];
      pa.q[0] = *(const uint4*)pp;
      pa.q[1] = *(const uint4*)(pp + 16);
    }
    // --- O += P(16x32) . V(32x64) ---
#pragma unroll
    for (int f = 0; f < 4; f++) {
      Frag vf;
      const unsigned short* vp = &Vs[f * 16 + m16][kb8];
      vf.q[0] = *(const uint4*)vp;
      vf.q[1] = *(const uint4*)(vp + 16);
      o[f] = __builtin_amdgcn_wmma_f32_16x16x32_bf16(
          false, pa.v, false, vf.v, (short)0, o[f], false, false);
    }
    __syncthreads();
  }

  // --- normalize + store ---
#pragma unroll
  for (int r = 0; r < 8; r++) {
    float inv = 1.0f / rl[r];
    int token = b * 1024 + qrow + r + 8 * hi;
#pragma unroll
    for (int f = 0; f < 4; f++) {
      int col = h * HD + f * 16 + m16;
      Obf[(size_t)token * DM + col] = (unsigned short)f2bfu(o[f][r] * inv);
    }
  }
}

// ---------------------------------------------------------------------------
// Host launch
// ---------------------------------------------------------------------------
extern "C" void kernel_launch(void* const* d_in, const int* in_sizes, int n_in,
                              void* d_out, int out_size, void* d_ws, size_t ws_size,
                              hipStream_t stream) {
  const float* x   = (const float*)d_in[0];
  const float* rw1 = (const float*)d_in[1];
  const float* rb1 = (const float*)d_in[2];
  const float* rw2 = (const float*)d_in[3];
  const float* rb2 = (const float*)d_in[4];
  const float* iw  = (const float*)d_in[5];
  const float* ib  = (const float*)d_in[6];
  const float* ow  = (const float*)d_in[7];
  const float* ob  = (const float*)d_in[8];
  const float* fw  = (const float*)d_in[9];
  const float* fb  = (const float*)d_in[10];

  char* ws = (char*)d_ws;
  auto al = [](size_t v) { return (v + 255) & ~(size_t)255; };
  size_t off = 0;
  unsigned short* x_bf  = (unsigned short*)(ws + off); off = al(off + (size_t)NT * DM * 2);
  unsigned short* h_bf  = (unsigned short*)(ws + off); off = al(off + (size_t)NT * 512 * 2);
  int* assign           = (int*)(ws + off);            off = al(off + (size_t)NT * 4);
  unsigned short* qkv   = (unsigned short*)(ws + off); off = al(off + (size_t)NT * 3 * DM * 2);
  unsigned short* o_bf  = (unsigned short*)(ws + off); off = al(off + (size_t)NT * DM * 2);
  float* combined       = (float*)(ws + off);          off = al(off + (size_t)NT * DM * 4);
  unsigned short* rw1_b = (unsigned short*)(ws + off); off = al(off + (size_t)512 * DM * 2);
  unsigned short* iw_b  = (unsigned short*)(ws + off); off = al(off + (size_t)NS * 3 * DM * DM * 2);
  unsigned short* ow_b  = (unsigned short*)(ws + off); off = al(off + (size_t)NS * DM * DM * 2);
  unsigned short* fw_b  = (unsigned short*)(ws + off); off = al(off + (size_t)DM * DM * 2);

  float* out = (float*)d_out;
  int* out_assign = (int*)(out + (size_t)NT * DM);

  // 0) one-time bf16 conversions: activations + all GEMM weights
  {
    int n4;
    n4 = NT * DM / 4;
    cvt_f32_bf16_kernel<<<(n4 + 255) / 256, 256, 0, stream>>>(x, x_bf, n4);
    n4 = 512 * DM / 4;
    cvt_f32_bf16_kernel<<<(n4 + 255) / 256, 256, 0, stream>>>(rw1, rw1_b, n4);
    n4 = NS * 3 * DM * DM / 4;
    cvt_f32_bf16_kernel<<<(n4 + 255) / 256, 256, 0, stream>>>(iw, iw_b, n4);
    n4 = NS * DM * DM / 4;
    cvt_f32_bf16_kernel<<<(n4 + 255) / 256, 256, 0, stream>>>(ow, ow_b, n4);
    n4 = DM * DM / 4;
    cvt_f32_bf16_kernel<<<(n4 + 255) / 256, 256, 0, stream>>>(fw, fw_b, n4);
  }

  // 1) router hidden: h = relu(x @ W1^T + b1)   [4096 x 512]
  gemm_bf16_wmma<MODE_BF16_RELU, false><<<dim3(512 / 128, NT / 128), 256, 0, stream>>>(
      x_bf, rw1_b, rb1, h_bf, nullptr, -1, NT, 512, DM);

  // 2) router logits + argmax dispatch (also emits assignments output)
  router_logits_kernel<<<NT / 256, 256, 0, stream>>>(h_bf, rw2, rb2, assign, out_assign);

  // 3) per-stratum masked MHA, accumulated into combined
  for (int s = 0; s < NS; s++) {
    gemm_bf16_wmma<MODE_BF16, true><<<dim3(3 * DM / 128, NT / 128), 256, 0, stream>>>(
        x_bf, iw_b + (size_t)s * 3 * DM * DM, ib + (size_t)s * 3 * DM,
        qkv, assign, s, NT, 3 * DM, DM);

    attn_wmma_kernel<<<dim3(4 * NH, 1024 / 128), 256, 0, stream>>>(qkv, o_bf);

    if (s == 0) {
      gemm_bf16_wmma<MODE_F32_SET_MASK, false><<<dim3(DM / 128, NT / 128), 256, 0, stream>>>(
          o_bf, ow_b + (size_t)s * DM * DM, ob + (size_t)s * DM,
          combined, assign, s, NT, DM, DM);
    } else {
      gemm_bf16_wmma<MODE_F32_ACC_MASK, false><<<dim3(DM / 128, NT / 128), 256, 0, stream>>>(
          o_bf, ow_b + (size_t)s * DM * DM, ob + (size_t)s * DM,
          combined, assign, s, NT, DM, DM);
    }
  }

  // 4) final projection: out = combined @ Wf^T + bf (reuse x_bf as staging)
  cvt_f32_bf16_kernel<<<(NT * DM / 4 + 255) / 256, 256, 0, stream>>>(combined, x_bf, NT * DM / 4);
  gemm_bf16_wmma<MODE_F32, false><<<dim3(DM / 128, NT / 128), 256, 0, stream>>>(
      x_bf, fw_b, fb, out, nullptr, -1, NT, DM, DM);
}
